// TorchKernel_84533546320034
// MI455X (gfx1250) — matrix operand order repaired and checked by reading, not compiled, not run on hardware
//
#include <hip/hip_runtime.h>

// ---------------------------------------------------------------------------
// RBF kernel matrix on MI455X (gfx1250, wave32, WMMA).
//   out[i,j] = exp(-GAMMA * max(||x_i||^2 + ||y_j||^2 - 2 x_i.y_j, 0))
// Cross term via v_wmma_f32_16x16x32_f16 (f16 in, f32 accumulate).
// Store-bound problem (268 MB output @ 23.3 TB/s ~ 11.5 us floor); inputs are
// L2-resident, so the kernel just streams WMMA tiles and coalesced stores.
// ---------------------------------------------------------------------------

typedef __attribute__((ext_vector_type(16))) _Float16     v16h;
typedef __attribute__((ext_vector_type(8)))  float        v8f;
typedef __attribute__((ext_vector_type(4)))  unsigned int u32x4;

#define GAMMA   0.015625f
#define NMAT    8192
#define DDIM    64
#define BT      128      // block tile edge (rows of X and rows of Y)
#define LDH     72       // padded LDS row stride in halfs (144 B, 16B-aligned)

union HFrag { v16h h; u32x4 q[2]; };

__global__ __launch_bounds__(256)
void rbf_wmma_kernel(const float* __restrict__ X,
                     const float* __restrict__ Y,
                     float* __restrict__ out) {
    __shared__ _Float16 As[BT][LDH];   // data tile, f16
    __shared__ _Float16 Bs[BT][LDH];   // support-vector tile, f16
    __shared__ float    Xsq[BT];       // row norms (f32, exact)
    __shared__ float    Ysq[BT];

    const int tid  = threadIdx.x;
    const int lane = tid & 31;
    const int wave = tid >> 5;          // 0..7
    const int l    = lane & 15;         // lane within half-wave
    const int hi   = lane >> 4;         // 0: lanes 0-15, 1: lanes 16-31

    const int rowBase = blockIdx.y * BT;
    const int colBase = blockIdx.x * BT;

    // ---- Stage: f32 -> f16 conversion into LDS + f32 row norms -----------
    {
        const bool doB = (tid >= BT);
        const int  r   = doB ? (tid - BT) : tid;
        const float4* src = (const float4*)(doB ? (Y + (size_t)(colBase + r) * DDIM)
                                                : (X + (size_t)(rowBase + r) * DDIM));
        _Float16* dst = doB ? &Bs[r][0] : &As[r][0];
        float ss = 0.0f;
#pragma unroll
        for (int i = 0; i < 16; i += 2) {
            float4 a = src[i];
            float4 b = src[i + 1];
            ss += a.x*a.x + a.y*a.y + a.z*a.z + a.w*a.w
                + b.x*b.x + b.y*b.y + b.z*b.z + b.w*b.w;
            union { _Float16 h[8]; u32x4 q; } p;
            p.h[0] = (_Float16)a.x; p.h[1] = (_Float16)a.y;
            p.h[2] = (_Float16)a.z; p.h[3] = (_Float16)a.w;
            p.h[4] = (_Float16)b.x; p.h[5] = (_Float16)b.y;
            p.h[6] = (_Float16)b.z; p.h[7] = (_Float16)b.w;
            *(u32x4*)(dst + i * 4) = p.q;   // ds_store_b128
        }
        if (doB) Ysq[r] = ss; else Xsq[r] = ss;
    }
    __syncthreads();

    // ---- Compute: each wave owns a 64x32 sub-tile = 4x2 WMMA tiles -------
    const int wr = (wave & 1) * 64;     // wave row offset within block tile
    const int wc = (wave >> 1) * 32;    // wave col offset within block tile

    v8f acc[4][2] = {};

#pragma unroll
    for (int kc = 0; kc < DDIM; kc += 32) {
        HFrag aF[4], bF[2];
        // A fragment, 16x32 f16 (ISA 7.12.2): lanes 0-15 hold K k..k+7 and
        // k+16..k+23; lanes 16-31 hold K k+8..k+15 and k+24..k+31.
#pragma unroll
        for (int rt = 0; rt < 4; ++rt) {
            const _Float16* ap = &As[wr + rt * 16 + l][0];
            aF[rt].q[0] = *(const u32x4*)(ap + kc + hi * 8);        // ds_load_b128
            aF[rt].q[1] = *(const u32x4*)(ap + kc + 16 + hi * 8);   // ds_load_b128
        }
        // B fragment, 32x16 f16: lane = column N; lanes 0-15 hold K k..k+7 and
        // k+16..k+23; lanes 16-31 hold K k+8..k+15 and k+24..k+31 (matches A).
#pragma unroll
        for (int ct = 0; ct < 2; ++ct) {
            const _Float16* bp = &Bs[wc + ct * 16 + l][0];
            bF[ct].q[0] = *(const u32x4*)(bp + kc + hi * 8);        // ds_load_b128
            bF[ct].q[1] = *(const u32x4*)(bp + kc + 16 + hi * 8);   // ds_load_b128
        }
#pragma unroll
        for (int rt = 0; rt < 4; ++rt)
#pragma unroll
            for (int ct = 0; ct < 2; ++ct)
                acc[rt][ct] = __builtin_amdgcn_wmma_f32_16x16x32_f16(
                    /*neg_a=*/false, aF[rt].h,
                    /*neg_b=*/false, bF[ct].h,
                    /*c_mod=*/(short)0, acc[rt][ct],
                    /*reuse_a=*/false, /*reuse_b=*/false);
    }

    // ---- Epilogue: d2 = x^2 + y^2 - 2*cross, out = exp(-gamma*d2) --------
    // C/D layout: VGPR v, lanes 0-15 -> M=v, N=lane; lanes 16-31 -> M=8+v.
#pragma unroll
    for (int rt = 0; rt < 4; ++rt) {
#pragma unroll
        for (int ct = 0; ct < 2; ++ct) {
            const int    n  = wc + ct * 16 + l;
            const float  ys = Ysq[n];
            const size_t oc = (size_t)(colBase + n);
#pragma unroll
            for (int v = 0; v < 8; ++v) {
                const int m  = wr + rt * 16 + hi * 8 + v;
                float d2 = Xsq[m] + ys - 2.0f * acc[rt][ct][v];
                d2 = fmaxf(d2, 0.0f);
                out[(size_t)(rowBase + m) * NMAT + oc] = __expf(-GAMMA * d2);
            }
        }
    }
}

extern "C" void kernel_launch(void* const* d_in, const int* in_sizes, int n_in,
                              void* d_out, int out_size, void* d_ws, size_t ws_size,
                              hipStream_t stream) {
    const float* X = (const float*)d_in[0];   // data            [8192, 64] f32
    const float* Y = (const float*)d_in[1];   // support_vectors [8192, 64] f32
    float* out = (float*)d_out;               // [8192, 8192] f32

    dim3 grid(NMAT / BT, NMAT / BT);          // 64 x 64 blocks
    rbf_wmma_kernel<<<grid, 256, 0, stream>>>(X, Y, out);
}
